// Transformer_16844861735767
// MI455X (gfx1250) — compile-verified, matrix operand order checked by reading, not run on hardware
//
#include <hip/hip_runtime.h>
#include <stdint.h>

// ---------------------------------------------------------------------------
// Transformer forward (B=4, T=1024, D=512, H=8 full-width heads, L=6, V=128)
// for gfx1250. GEMMs: v_wmma_f32_16x16x32_f16, operands pre-converted to f16
// in global memory, tiles staged LDS-side by the Tensor Data Mover
// (tensor_load_to_lds, TENSORcnt-tracked, double-buffered).
// ---------------------------------------------------------------------------

typedef __attribute__((ext_vector_type(16))) _Float16 v16h;
typedef __attribute__((ext_vector_type(8)))  float    v8f;
typedef __attribute__((ext_vector_type(4)))  unsigned int u32x4;
typedef __attribute__((ext_vector_type(8)))  unsigned int u32x8;

#define TB   4
#define TT   1024
#define TD   512
#define TH   8
#define TL   6
#define TV   128
#define TDFF 2048
#define TBT  4096

#define LDS_STR 40   // padded LDS row stride in halves (64B data + 16B pad)

// ---------------------------------------------------------------------------
// helpers
// ---------------------------------------------------------------------------
__device__ __forceinline__ uint32_t lds_addr_of(const void* p) {
    // Flat LDS addresses are {aperture, offset}: low 32 bits are the LDS
    // byte address the TDM descriptor wants.
    return (uint32_t)(uintptr_t)p;
}

__device__ __forceinline__ v16h ld_frag(const _Float16* p) {
    // halves [0..7] -> K = klo..klo+7, halves [8..15] -> K = klo+16..klo+23
    // (CDNA5 16-bit A/B operand VGPR layout), two contiguous 16B LDS reads.
    v16h f;
    reinterpret_cast<uint4*>(&f)[0] = *reinterpret_cast<const uint4*>(p);
    reinterpret_cast<uint4*>(&f)[1] = *reinterpret_cast<const uint4*>(p + 16);
    return f;
}

__device__ __forceinline__ v8f wmma_step(v16h a, v16h b, v8f c) {
    return __builtin_amdgcn_wmma_f32_16x16x32_f16(
        false, a, false, b, (short)0, c, false, false);
}

// Issue one 2-D TDM tile load: tile0 x tile1 elements of 2B data from a
// row-major [dim1][stride0] tensor into LDS at lds_addr, padding 16B after
// every 64B row so LDS rows land on an 80B (LDS_STR) stride.
__device__ __forceinline__ void tdm_load_2d(uint32_t lds_addr, const _Float16* gp,
                                            uint32_t tdim0, uint32_t tdim1,
                                            uint32_t stride0,
                                            uint32_t tile0, uint32_t tile1) {
    uint64_t ga = (uint64_t)(uintptr_t)gp;
    u32x4 g0;
    g0[0] = 1u;                                  // count=1, user descriptor
    g0[1] = lds_addr;                            // lds_addr (bytes)
    g0[2] = (uint32_t)ga;                        // global_addr[31:0]
    g0[3] = (uint32_t)(ga >> 32) | (2u << 30);   // global_addr[56:32], type=2
    u32x8 g1;
    // data_size=1 (2B) | pad_enable | pad_interval=3 (16 DW) | pad_amount=3 (4 DW)
    g1[0] = (1u << 16) | (1u << 20) | (3u << 22) | (3u << 25);
    g1[1] = (tdim0 & 0xFFFFu) << 16;                     // tensor_dim0[15:0]
    g1[2] = (tdim0 >> 16) | ((tdim1 & 0xFFFFu) << 16);   // dim0[31:16] dim1[15:0]
    g1[3] = (tdim1 >> 16) | (tile0 << 16);               // dim1[31:16] tile_dim0
    g1[4] = tile1 & 0xFFFFu;                             // tile_dim1 (tile_dim2=0)
    g1[5] = stride0;                                     // tensor_dim0_stride[31:0]
    g1[6] = 0u;
    g1[7] = 0u;
    asm volatile("tensor_load_to_lds %0, %1" :: "s"(g0), "s"(g1) : "memory");
}

// ---------------------------------------------------------------------------
// GEMM: C[M,N] = A[M,K] * B^T  with A f16 [M][K] (lda), B f16 [N][K] (ldb).
// Outputs: Cf (f32, nullable) and/or Ch (f16, nullable), bias[n] (nullable),
// optional relu. Block tile 128x128, 8 waves (2x4), wave tile 64x32
// (4x2 WMMA frags). Tiles staged by TDM, double-buffered.
// ---------------------------------------------------------------------------
__global__ __launch_bounds__(256)
void gemm_tdm_wmma(const _Float16* __restrict__ A, const _Float16* __restrict__ B,
                   float* __restrict__ Cf, _Float16* __restrict__ Ch,
                   const float* __restrict__ bias,
                   int M, int N, int K, int lda, int ldb, int ldc,
                   long long sA, long long sB, long long sC, int relu) {
    __shared__ __align__(16) _Float16 As[2][128 * LDS_STR];
    __shared__ __align__(16) _Float16 Bs[2][128 * LDS_STR];

    const int batch = blockIdx.z;
    A += (size_t)batch * sA;
    B += (size_t)batch * sB;

    const int bm = blockIdx.y * 128;
    const int bn = blockIdx.x * 128;

    const int tid  = threadIdx.x;
    const int lane = tid & 31;
    const int wave = tid >> 5;
    const int wm   = (wave & 1) * 64;
    const int wn   = (wave >> 1) * 32;
    const int lr   = lane & 15;
    const int hi   = lane >> 4;
    const int klo  = hi * 8;

    v8f acc[4][2] = {};

    if (wave == 0) {
        tdm_load_2d(lds_addr_of(&As[0][0]), A + (size_t)bm * lda,
                    (uint32_t)K, (uint32_t)(M - bm), (uint32_t)lda, 32, 128);
        tdm_load_2d(lds_addr_of(&Bs[0][0]), B + (size_t)bn * ldb,
                    (uint32_t)K, (uint32_t)(N - bn), (uint32_t)ldb, 32, 128);
        __builtin_amdgcn_s_wait_tensorcnt(0);
    }
    __syncthreads();

    int buf = 0;
    for (int k0 = 0; k0 < K; k0 += 32) {
        // TDM prefetch of next K-panel into the other buffer
        if (wave == 0 && (k0 + 32) < K) {
            tdm_load_2d(lds_addr_of(&As[buf ^ 1][0]),
                        A + (size_t)bm * lda + (k0 + 32),
                        (uint32_t)(K - k0 - 32), (uint32_t)(M - bm),
                        (uint32_t)lda, 32, 128);
            tdm_load_2d(lds_addr_of(&Bs[buf ^ 1][0]),
                        B + (size_t)bn * ldb + (k0 + 32),
                        (uint32_t)(K - k0 - 32), (uint32_t)(N - bn),
                        (uint32_t)ldb, 32, 128);
        }

        v16h af[4], bf[2];
        #pragma unroll
        for (int i = 0; i < 4; ++i)
            af[i] = ld_frag(&As[buf][(wm + i * 16 + lr) * LDS_STR + klo]);
        #pragma unroll
        for (int j = 0; j < 2; ++j)
            bf[j] = ld_frag(&Bs[buf][(wn + j * 16 + lr) * LDS_STR + klo]);

        #pragma unroll
        for (int i = 0; i < 4; ++i)
            #pragma unroll
            for (int j = 0; j < 2; ++j)
                acc[i][j] = wmma_step(af[i], bf[j], acc[i][j]);

        if (wave == 0) __builtin_amdgcn_s_wait_tensorcnt(0);
        __syncthreads();
        buf ^= 1;
    }

    // epilogue: bias (+relu), f32 and/or f16 stores
    const int colb = bn + wn + lr;
    const int rowb = bm + wm + hi * 8;
    #pragma unroll
    for (int i = 0; i < 4; ++i) {
        #pragma unroll
        for (int j = 0; j < 2; ++j) {
            const int col = colb + j * 16;
            const float bb = bias ? bias[col] : 0.f;
            #pragma unroll
            for (int r = 0; r < 8; ++r) {
                const int row = rowb + i * 16 + r;
                float v = acc[i][j][r] + bb;
                if (relu) v = fmaxf(v, 0.f);
                const size_t idx = (size_t)batch * sC + (size_t)row * ldc + col;
                if (Cf) Cf[idx] = v;
                if (Ch) Ch[idx] = (_Float16)v;
            }
        }
    }
}

// ---------------------------------------------------------------------------
// f32 [R][C] -> f16 [C][R] transposed convert (batched over blockIdx.z)
// ---------------------------------------------------------------------------
__global__ __launch_bounds__(256)
void convert_tr_f16(const float* __restrict__ in, _Float16* __restrict__ out,
                    int R, int C) {
    const long long nm = (long long)R * C;
    const float* src = in  + (size_t)blockIdx.z * nm;
    _Float16*    dst = out + (size_t)blockIdx.z * nm;
    long long i = (long long)blockIdx.x * 256 + threadIdx.x;
    if (i < nm) {
        int r = (int)(i / C), c = (int)(i % C);
        dst[(size_t)c * R + r] = (_Float16)src[i];
    }
}

// f16 [R][C] -> f16 [C][R] transpose (batched over blockIdx.z)
__global__ __launch_bounds__(256)
void transpose_f16(const _Float16* __restrict__ in, _Float16* __restrict__ out,
                   int R, int C) {
    const long long nm = (long long)R * C;
    const _Float16* src = in  + (size_t)blockIdx.z * nm;
    _Float16*       dst = out + (size_t)blockIdx.z * nm;
    long long i = (long long)blockIdx.x * 256 + threadIdx.x;
    if (i < nm) {
        int r = (int)(i / C), c = (int)(i % C);
        dst[(size_t)c * R + r] = src[i];
    }
}

// ---------------------------------------------------------------------------
// Embedding gather -> f32 residual stream + f16 GEMM operand copy
// ---------------------------------------------------------------------------
__global__ __launch_bounds__(256)
void embed_kernel(const int* __restrict__ tokens, const float* __restrict__ emb,
                  float* __restrict__ x32, _Float16* __restrict__ x16,
                  long long n) {
    long long i = (long long)blockIdx.x * 256 + threadIdx.x;
    if (i < n) {
        int tok = tokens[i / TD];
        float v = emb[(size_t)tok * TD + (i % TD)];
        x32[i] = v;
        x16[i] = (_Float16)v;
    }
}

// ---------------------------------------------------------------------------
// Causal softmax: f32 scores row -> f16 probability row (no scaling, faithful)
// ---------------------------------------------------------------------------
__global__ __launch_bounds__(256)
void softmax_causal(const float* __restrict__ scores, _Float16* __restrict__ probs) {
    const int t = blockIdx.x;
    const int b = blockIdx.y;
    const float* row = scores + ((size_t)b * TT + t) * TT;
    _Float16*    out = probs  + ((size_t)b * TT + t) * TT;
    __shared__ float red[256];
    const int tid = threadIdx.x;
    const float NEG = -__builtin_inff();

    float m = NEG;
    for (int s = tid; s < TT; s += 256) m = fmaxf(m, (s <= t) ? row[s] : NEG);
    red[tid] = m; __syncthreads();
    for (int o = 128; o > 0; o >>= 1) {
        if (tid < o) red[tid] = fmaxf(red[tid], red[tid + o]);
        __syncthreads();
    }
    const float rmax = red[0]; __syncthreads();

    float sum = 0.f;
    for (int s = tid; s < TT; s += 256)
        sum += (s <= t) ? __expf(row[s] - rmax) : 0.f;
    red[tid] = sum; __syncthreads();
    for (int o = 128; o > 0; o >>= 1) {
        if (tid < o) red[tid] += red[tid + o];
        __syncthreads();
    }
    const float inv = 1.f / red[0];

    for (int s = tid; s < TT; s += 256)
        out[s] = (_Float16)((s <= t) ? __expf(row[s] - rmax) * inv : 0.f);
}

// ---------------------------------------------------------------------------
// LayerNorm(a + b) * g + beta -> f32 out (nullable) and f16 out (nullable)
// ---------------------------------------------------------------------------
__global__ __launch_bounds__(256)
void add_layernorm(const float* __restrict__ a, const float* __restrict__ b,
                   const float* __restrict__ g, const float* __restrict__ beta,
                   float* __restrict__ out32, _Float16* __restrict__ out16) {
    const size_t base = (size_t)blockIdx.x * TD;
    const int tid = threadIdx.x;
    __shared__ float red[256];

    float s = 0.f;
    for (int e = tid; e < TD; e += 256)
        s += a[base + e] + (b ? b[base + e] : 0.f);
    red[tid] = s; __syncthreads();
    for (int o = 128; o > 0; o >>= 1) {
        if (tid < o) red[tid] += red[tid + o];
        __syncthreads();
    }
    const float mean = red[0] * (1.f / TD); __syncthreads();

    float vs = 0.f;
    for (int e = tid; e < TD; e += 256) {
        float xv = a[base + e] + (b ? b[base + e] : 0.f);
        float d  = xv - mean;
        vs += d * d;
    }
    red[tid] = vs; __syncthreads();
    for (int o = 128; o > 0; o >>= 1) {
        if (tid < o) red[tid] += red[tid + o];
        __syncthreads();
    }
    const float rstd = rsqrtf(red[0] * (1.f / TD) + 1e-5f);

    for (int e = tid; e < TD; e += 256) {
        float xv = a[base + e] + (b ? b[base + e] : 0.f);
        float y  = (xv - mean) * rstd * g[e] + beta[e];
        if (out32) out32[base + e] = y;
        if (out16) out16[base + e] = (_Float16)y;
    }
}

// ---------------------------------------------------------------------------
// Cross-entropy: loss += -log_softmax(logits[row])[label[row]] * scale
// ---------------------------------------------------------------------------
__global__ __launch_bounds__(128)
void ce_loss(const float* __restrict__ logits, const int* __restrict__ labels,
             float* __restrict__ loss, float scale) {
    const size_t base = (size_t)blockIdx.x * TV;
    const int tid = threadIdx.x;
    __shared__ float red[128];

    const float x = logits[base + tid];
    red[tid] = x; __syncthreads();
    for (int o = 64; o > 0; o >>= 1) {
        if (tid < o) red[tid] = fmaxf(red[tid], red[tid + o]);
        __syncthreads();
    }
    const float m = red[0]; __syncthreads();

    red[tid] = __expf(x - m); __syncthreads();
    for (int o = 64; o > 0; o >>= 1) {
        if (tid < o) red[tid] += red[tid + o];
        __syncthreads();
    }
    const float lse = m + __logf(red[0]);

    if (tid == labels[blockIdx.x]) atomicAdd(loss, (lse - x) * scale);
}

// ---------------------------------------------------------------------------
// Orchestration
// ---------------------------------------------------------------------------
extern "C" void kernel_launch(void* const* d_in, const int* in_sizes, int n_in,
                              void* d_out, int out_size, void* d_ws, size_t ws_size,
                              hipStream_t stream) {
    (void)in_sizes; (void)n_in; (void)out_size; (void)ws_size;

    const int*   tokens = (const int*)  d_in[0];
    const int*   labels = (const int*)  d_in[1];
    const float* emb    = (const float*)d_in[2];
    const float* Wq     = (const float*)d_in[3];
    const float* bq     = (const float*)d_in[4];
    const float* Wk     = (const float*)d_in[5];
    const float* bk     = (const float*)d_in[6];
    const float* Wv     = (const float*)d_in[7];
    const float* bv     = (const float*)d_in[8];
    const float* Wo     = (const float*)d_in[9];
    const float* bo     = (const float*)d_in[10];
    const float* ln1_g  = (const float*)d_in[11];
    const float* ln1_b  = (const float*)d_in[12];
    const float* W1     = (const float*)d_in[13];
    const float* b1     = (const float*)d_in[14];
    const float* W2     = (const float*)d_in[15];
    const float* b2     = (const float*)d_in[16];
    const float* ln2_g  = (const float*)d_in[17];
    const float* ln2_b  = (const float*)d_in[18];
    const float* lnf_g  = (const float*)d_in[19];
    const float* lnf_b  = (const float*)d_in[20];
    const float* Wout   = (const float*)d_in[21];
    const float* bout   = (const float*)d_in[22];

    // ---- workspace carve-up ----
    char* wp = (char*)d_ws;
    auto carve = [&](size_t bytes) -> void* {
        void* r = (void*)wp;
        wp += (bytes + 255) & ~(size_t)255;
        return r;
    };
    // f32 buffers
    float* xa     = (float*)carve((size_t)TBT * TD * 4);
    float* xb     = (float*)carve((size_t)TBT * TD * 4);
    float* proj   = (float*)carve((size_t)TBT * TD * 4);
    float* scores = (float*)carve((size_t)TB * TT * TT * 4);
    // f16 activation buffers
    _Float16* xh    = (_Float16*)carve((size_t)TBT * TD * 2);
    _Float16* qh    = (_Float16*)carve((size_t)TBT * TD * 2);
    _Float16* kh    = (_Float16*)carve((size_t)TBT * TD * 2);
    _Float16* vh    = (_Float16*)carve((size_t)TBT * TD * 2);
    _Float16* vt    = (_Float16*)carve((size_t)TBT * TD * 2);
    _Float16* probs = (_Float16*)carve((size_t)TB * TT * TT * 2);
    _Float16* oc    = (_Float16*)carve((size_t)TBT * TH * TD * 2);
    _Float16* hid   = (_Float16*)carve((size_t)TBT * TDFF * 2);
    // f16 transposed weights ([N][K] so every GEMM B-operand is row-contiguous)
    _Float16* wqt   = (_Float16*)carve((size_t)TL * TH * TD * TD * 2);
    _Float16* wkt   = (_Float16*)carve((size_t)TL * TH * TD * TD * 2);
    _Float16* wvt   = (_Float16*)carve((size_t)TL * TH * TD * TD * 2);
    _Float16* wot   = (_Float16*)carve((size_t)TL * TD * TH * TD * 2);
    _Float16* w1t   = (_Float16*)carve((size_t)TL * TD * TDFF * 2);
    _Float16* w2t   = (_Float16*)carve((size_t)TL * TDFF * TD * 2);
    _Float16* woutt = (_Float16*)carve((size_t)TD * TV * 2);

    auto gemm = [&](const _Float16* A, const _Float16* Bm,
                    float* Cf, _Float16* Ch, const float* bias,
                    int M, int N, int K, int lda, int ldb, int ldc,
                    long long sA, long long sB, long long sC,
                    int batch, int relu) {
        dim3 grid(N / 128, M / 128, batch);
        gemm_tdm_wmma<<<grid, 256, 0, stream>>>(A, Bm, Cf, Ch, bias, M, N, K,
                                                lda, ldb, ldc, sA, sB, sC, relu);
    };
    auto convT = [&](const float* in, _Float16* out, int R, int C, int batch) {
        long long nm = (long long)R * C;
        dim3 grid((unsigned)((nm + 255) / 256), 1, batch);
        convert_tr_f16<<<grid, 256, 0, stream>>>(in, out, R, C);
    };

    // ---- one-time (per call) weight convert+transpose to f16 ----
    convT(Wq,   wqt,   TD,      TD,   TL * TH);
    convT(Wk,   wkt,   TD,      TD,   TL * TH);
    convT(Wv,   wvt,   TD,      TD,   TL * TH);
    convT(Wo,   wot,   TH * TD, TD,   TL);
    convT(W1,   w1t,   TD,      TDFF, TL);
    convT(W2,   w2t,   TDFF,    TD,   TL);
    convT(Wout, woutt, TD,      TV,   1);

    // ---- embedding ----
    {
        long long n = (long long)TBT * TD;
        embed_kernel<<<(unsigned)((n + 255) / 256), 256, 0, stream>>>(
            tokens, emb, xa, xh, n);
    }

    for (int l = 0; l < TL; ++l) {
        for (int h = 0; h < TH; ++h) {
            const size_t wi = (size_t)(l * TH + h);
            // Q/K/V projections (f16 out only)
            gemm(xh, wqt + wi * TD * TD, nullptr, qh, bq + wi * TD,
                 TBT, TD, TD, TD, TD, TD, 0, 0, 0, 1, 0);
            gemm(xh, wkt + wi * TD * TD, nullptr, kh, bk + wi * TD,
                 TBT, TD, TD, TD, TD, TD, 0, 0, 0, 1, 0);
            gemm(xh, wvt + wi * TD * TD, nullptr, vh, bv + wi * TD,
                 TBT, TD, TD, TD, TD, TD, 0, 0, 0, 1, 0);
            // V -> V^T per batch so attn GEMM's B-operand is [N=D][K=T]
            {
                long long nm = (long long)TT * TD;
                dim3 g((unsigned)((nm + 255) / 256), 1, TB);
                transpose_f16<<<g, 256, 0, stream>>>(vh, vt, TT, TD);
            }
            // scores = Q * K^T (K already [T][D] = [N][K]), batched over B
            gemm(qh, kh, scores, nullptr, nullptr,
                 TT, TT, TD, TD, TD, TT,
                 (long long)TT * TD, (long long)TT * TD, (long long)TT * TT,
                 TB, 0);
            softmax_causal<<<dim3(TT, TB), 256, 0, stream>>>(scores, probs);
            // oc[:, h*D:(h+1)*D] = P * V, batched over B
            gemm(probs, vt, nullptr, oc + (size_t)h * TD, nullptr,
                 TT, TD, TT, TT, TT, TH * TD,
                 (long long)TT * TT, (long long)TD * TT, (long long)TT * TH * TD,
                 TB, 0);
        }
        // output projection
        gemm(oc, wot + (size_t)l * TD * TH * TD, proj, nullptr, bo + (size_t)l * TD,
             TBT, TD, TH * TD, TH * TD, TH * TD, TD, 0, 0, 0, 1, 0);
        add_layernorm<<<TBT, 256, 0, stream>>>(xa, proj,
                                               ln1_g + (size_t)l * TD,
                                               ln1_b + (size_t)l * TD, xb, xh);
        // MLP
        gemm(xh, w1t + (size_t)l * TD * TDFF, nullptr, hid, b1 + (size_t)l * TDFF,
             TBT, TDFF, TD, TD, TD, TDFF, 0, 0, 0, 1, 1 /*relu*/);
        gemm(hid, w2t + (size_t)l * TDFF * TD, proj, nullptr, b2 + (size_t)l * TD,
             TBT, TD, TDFF, TDFF, TDFF, TD, 0, 0, 0, 1, 0);
        add_layernorm<<<TBT, 256, 0, stream>>>(xb, proj,
                                               ln2_g + (size_t)l * TD,
                                               ln2_b + (size_t)l * TD, xa, xh);
    }

    // ---- final LN + logits + loss ----
    add_layernorm<<<TBT, 256, 0, stream>>>(xa, nullptr, lnf_g, lnf_b, nullptr, xh);

    float* logits = (float*)d_out;
    gemm(xh, woutt, logits, nullptr, bout, TBT, TV, TD, TD, TD, TV, 0, 0, 0, 1, 0);

    float* loss = logits + (size_t)TBT * TV;
    hipMemsetAsync(loss, 0, sizeof(float), stream);
    ce_loss<<<TBT, 128, 0, stream>>>(logits, labels, loss, 1.f / (float)TBT);
}
